// SPDNetBaseline_23433341567255
// MI455X (gfx1250) — matrix operand order repaired and checked by reading, not compiled, not run on hardware
//
#include <hip/hip_runtime.h>
#include <hip/hip_bf16.h>
#include <cstdint>

typedef __attribute__((ext_vector_type(16))) __bf16    v16bf;
typedef __attribute__((ext_vector_type(8)))  float     v8f;
typedef __attribute__((ext_vector_type(4)))  unsigned  v4u;

#define SPD_EPS 1e-3f
#define RE_EPS  1e-4f
#define B_DIM 512
#define C_DIM 128
#define T_DIM 1000
#define CHUNK 32
#define LDX   40   // LDS row stride (ushorts): 80B rows -> 16B-aligned, conflict-free

__device__ __forceinline__ unsigned short f2bf(float f) {
  unsigned u = __builtin_bit_cast(unsigned, f);
  u += 0x7FFFu + ((u >> 16) & 1u);          // round-to-nearest-even
  return (unsigned short)(u >> 16);
}
__device__ __forceinline__ float bf2f(unsigned short h) {
  unsigned u = ((unsigned)h) << 16;
  return __builtin_bit_cast(float, u);
}

__device__ __forceinline__ v8f wmma_bf16(v16bf a, v16bf b, v8f c) {
  return __builtin_amdgcn_wmma_f32_16x16x32_bf16(false, a, false, b, (short)0, c,
                                                 false, false);
}

// A fragment (16x32 bf16): rows m0..m0+15, K = k0..k0+31 from row-major LDS.
// Lane's 8 dwords live in two contiguous 16B groups -> 2x ds_load_b128.
// Requires (m*ld + kb) % 8 == 0 (ushorts): ld % 8 == 0 and k0 % 32 == 0.
__device__ __forceinline__ v16bf frag_A(const unsigned short* sh, int ld, int m0,
                                        int k0, int lane) {
  int m = m0 + (lane & 15);
  int kb = k0 + ((lane < 16) ? 0 : 8);
  const v4u* p = (const v4u*)(sh + m * ld + kb);
  union { v16bf v; v4u q[2]; } r;
  r.q[0] = p[0];   // dwords 0..3   (K = kb+0..7)
  r.q[1] = p[2];   // dwords 8..11  (K = kb+16..23)
  return r.v;
}

// B fragment (32x16 bf16) where B[k][n] = src[n][k] (src row = output column).
// Lane's 8 dwords live in two contiguous 16B groups (K = kb+0..7, kb+16..23) -> 2x ds_load_b128.
__device__ __forceinline__ v16bf frag_BT(const unsigned short* sh, int ld, int n0,
                                         int k0, int lane) {
  int n = n0 + (lane & 15);
  int kb = k0 + ((lane < 16) ? 0 : 8);
  const v4u* p = (const v4u*)(sh + n * ld + kb);
  union { v16bf v; v4u q[2]; } r;
  r.q[0] = p[0];
  r.q[1] = p[2];
  return r.v;
}

// ---------------------------------------------------------------------------
// Kernel 1: per-trial covariance via bf16 hi/lo-split WMMA (streaming, 1 WG/batch)
// cov = (X X^T - T mu mu^T) / (T-1) + eps*I, mean folded in algebraically.
// Double-buffered LDS planes: exactly one barrier per chunk; next chunk's
// global loads issue before the WMMA block so HBM latency overlaps compute.
// ---------------------------------------------------------------------------
__global__ __launch_bounds__(256) void cov_kernel(const float* __restrict__ x,
                                                  float* __restrict__ cov) {
  __shared__ unsigned short shHi[2][C_DIM * LDX];
  __shared__ unsigned short shLo[2][C_DIM * LDX];
  __shared__ float shSum[256];
  __shared__ float shMu[C_DIM];

  const int b    = blockIdx.x;
  const int tid  = threadIdx.x;
  const int lane = tid & 31;
  const int wave = tid >> 5;
  const float* xb = x + (size_t)b * C_DIM * T_DIM;

  v8f acc[8];
#pragma unroll
  for (int nt = 0; nt < 8; ++nt)
#pragma unroll
    for (int i = 0; i < 8; ++i) acc[nt][i] = 0.0f;

  const int row = tid >> 1;           // channel 0..127 (2 threads per row)
  const int tof = (tid & 1) * 16;     // t sub-offset within chunk
  const float* xrow = xb + (size_t)row * T_DIM;
  float rowsum = 0.0f;

  const int nch = (T_DIM + CHUNK - 1) / CHUNK;  // 32 (last chunk zero-padded)

  float v[16];
#pragma unroll
  for (int i = 0; i < 16; ++i) v[i] = xrow[tof + i];   // preload chunk 0

  for (int ch = 0; ch < nch; ++ch) {
    const int cur = ch & 1;
    unsigned short* hi = shHi[cur];
    unsigned short* lo = shLo[cur];
#pragma unroll
    for (int i = 0; i < 16; ++i) {
      rowsum += v[i];
      unsigned short h = f2bf(v[i]);
      hi[row * LDX + tof + i] = h;
      lo[row * LDX + tof + i] = f2bf(v[i] - bf2f(h));
    }
    __syncthreads();   // single barrier/chunk: buffer reuse distance spans it

    if (ch + 2 < nch)  // speculative prefetch 2 chunks ahead
      __builtin_prefetch(xrow + (ch + 2) * CHUNK + tof, 0, 1);

    if (ch + 1 < nch) {  // issue next chunk's loads before the WMMA block
      const int t0 = (ch + 1) * CHUNK + tof;
#pragma unroll
      for (int i = 0; i < 16; ++i) {
        int t = t0 + i;
        v[i] = (t < T_DIM) ? xrow[t] : 0.0f;
      }
    }

    v16bf ahi = frag_A(hi, LDX, wave * 16, 0, lane);
    v16bf alo = frag_A(lo, LDX, wave * 16, 0, lane);
#pragma unroll
    for (int nt = 0; nt < 8; ++nt) {
      v16bf bhi = frag_BT(hi, LDX, nt * 16, 0, lane);  // Gram: B = X^T
      v16bf blo = frag_BT(lo, LDX, nt * 16, 0, lane);
      acc[nt] = wmma_bf16(ahi, bhi, acc[nt]);
      acc[nt] = wmma_bf16(ahi, blo, acc[nt]);
      acc[nt] = wmma_bf16(alo, bhi, acc[nt]);
    }
  }

  __syncthreads();
  shSum[tid] = rowsum;
  __syncthreads();
  if (tid < C_DIM) shMu[tid] = (shSum[2 * tid] + shSum[2 * tid + 1]) * (1.0f / T_DIM);
  __syncthreads();

  const float inv = 1.0f / (float)(T_DIM - 1);
  const int n_in_tile = lane & 15;
  const int m_half = 8 * (lane >> 4);
#pragma unroll
  for (int nt = 0; nt < 8; ++nt) {
#pragma unroll
    for (int i = 0; i < 8; ++i) {
      int m = wave * 16 + i + m_half;
      int n = nt * 16 + n_in_tile;
      float cv = (acc[nt][i] - (float)T_DIM * shMu[m] * shMu[n]) * inv;
      if (m == n) cv += SPD_EPS;
      cov[((size_t)b * C_DIM + m) * C_DIM + n] = cv;
    }
  }
}

// ---------------------------------------------------------------------------
// Kernel 2: BiMap  S_out = W * S * W^T  (S symmetric), all WMMA, 1 WG/batch
// ---------------------------------------------------------------------------
template <int DIN, int DOUT>
__global__ __launch_bounds__(256) void bimap_kernel(const float* __restrict__ Sin,
                                                    const float* __restrict__ W,
                                                    float* __restrict__ Sout) {
  constexpr int LDA = DIN + 8;  // 16B-aligned rows, conflict-free frag loads
  __shared__ unsigned short shS[DIN * LDA];
  __shared__ unsigned short shW[DOUT * LDA];
  __shared__ unsigned short shP[DOUT * LDA];

  const int b    = blockIdx.x;
  const int tid  = threadIdx.x;
  const int lane = tid & 31;
  const int wave = tid >> 5;
  const float* S = Sin + (size_t)b * DIN * DIN;

  for (int idx = tid; idx < DIN * DIN; idx += 256)
    shS[(idx / DIN) * LDA + (idx % DIN)] = f2bf(S[idx]);
  for (int idx = tid; idx < DOUT * DIN; idx += 256)
    shW[(idx / DIN) * LDA + (idx % DIN)] = f2bf(W[idx]);
  __syncthreads();

  // Phase 1: P = W * S   (DOUT x DIN); B from shS rows (S symmetric)
  constexpr int NT1 = DIN / 16, KC1 = DIN / 32, TILES1 = (DOUT / 16) * NT1;
  for (int t = wave; t < TILES1; t += 8) {
    int mt = t / NT1, nt = t % NT1;
    v8f acc;
#pragma unroll
    for (int i = 0; i < 8; ++i) acc[i] = 0.0f;
#pragma unroll
    for (int kc = 0; kc < KC1; ++kc) {
      v16bf a  = frag_A (shW, LDA, mt * 16, kc * 32, lane);
      v16bf bb = frag_BT(shS, LDA, nt * 16, kc * 32, lane);
      acc = wmma_bf16(a, bb, acc);
    }
    int n = nt * 16 + (lane & 15);
#pragma unroll
    for (int i = 0; i < 8; ++i) {
      int m = mt * 16 + i + 8 * (lane >> 4);
      shP[m * LDA + n] = f2bf(acc[i]);
    }
  }
  __syncthreads();

  // Phase 2: S_out = P * W^T  (DOUT x DOUT); B[k][n] = W[n][k] -> shW rows
  constexpr int NT2 = DOUT / 16, KC2 = DIN / 32, TILES2 = (DOUT / 16) * NT2;
  float* out = Sout + (size_t)b * DOUT * DOUT;
  for (int t = wave; t < TILES2; t += 8) {
    int mt = t / NT2, nt = t % NT2;
    v8f acc;
#pragma unroll
    for (int i = 0; i < 8; ++i) acc[i] = 0.0f;
#pragma unroll
    for (int kc = 0; kc < KC2; ++kc) {
      v16bf a  = frag_A (shP, LDA, mt * 16, kc * 32, lane);
      v16bf bb = frag_BT(shW, LDA, nt * 16, kc * 32, lane);
      acc = wmma_bf16(a, bb, acc);
    }
    int n = nt * 16 + (lane & 15);
#pragma unroll
    for (int i = 0; i < 8; ++i) {
      int m = mt * 16 + i + 8 * (lane >> 4);
      out[(size_t)m * DOUT + n] = acc[i];
    }
  }
}

// ---------------------------------------------------------------------------
// Kernel 3: batched symmetric eig via Brent-Luk parallel cyclic Jacobi in LDS,
// then reconstruct V f(Lambda) V^T.  f = clamp (ReEig) or log(clamp) (fused
// ReEig+LogEig, mathematically identical to the reference's two-step form).
// Disjoint-pair rotations commute, so round-wide col-then-row update == J^T A J.
// ---------------------------------------------------------------------------
template <int D, bool TAKELOG>
__global__ __launch_bounds__(256) void jacobi_kernel(const float* __restrict__ Sin,
                                                     float* __restrict__ Sout,
                                                     int sweeps) {
  constexpr int LDA = D + 1;
  constexpr int NP  = D / 2;          // pairs per round
  constexpr int NG  = 256 / D;        // pair-groups per row-thread
  __shared__ float A[D * LDA];
  __shared__ float V[D * LDA];
  __shared__ float cc[NP], ss[NP];
  __shared__ int   pp[NP], qq[NP];
  __shared__ float fw[D];

  const int b   = blockIdx.x;
  const int tid = threadIdx.x;
  const float* S = Sin + (size_t)b * D * D;

  for (int idx = tid; idx < D * D; idx += 256) {
    int i = idx / D, j = idx % D;
    A[i * LDA + j] = S[idx];
    V[i * LDA + j] = (i == j) ? 1.0f : 0.0f;
  }
  __syncthreads();

  for (int sw = 0; sw < sweeps; ++sw) {
    for (int r = 0; r < D - 1; ++r) {
      if (tid < NP) {  // round-robin (circle method) pairing
        int k = tid, i, j;
        if (k == 0) { i = 0; j = 1 + ((r + D - 2) % (D - 1)); }
        else {
          i = 1 + ((r + k - 1) % (D - 1));
          j = 1 + ((r + D - 2 - k) % (D - 1));
        }
        int p = (i < j) ? i : j, q = (i < j) ? j : i;
        float app = A[p * LDA + p], aqq = A[q * LDA + q], apq = A[p * LDA + q];
        float c = 1.0f, s = 0.0f;
        if (fabsf(apq) > 1e-12f) {
          float tau = (aqq - app) / (2.0f * apq);
          float tt  = ((tau >= 0.0f) ? 1.0f : -1.0f) /
                      (fabsf(tau) + sqrtf(1.0f + tau * tau));
          c = rsqrtf(1.0f + tt * tt);
          s = tt * c;
        }
        cc[k] = c; ss[k] = s; pp[k] = p; qq[k] = q;
      }
      __syncthreads();
      {  // column rotations: A <- A*J, V <- V*J (disjoint column pairs)
        const int i = tid & (D - 1);
        for (int k = tid / D; k < NP; k += NG) {
          int p = pp[k], q = qq[k];
          float c = cc[k], s = ss[k];
          float aip = A[i * LDA + p], aiq = A[i * LDA + q];
          A[i * LDA + p] = c * aip - s * aiq;
          A[i * LDA + q] = s * aip + c * aiq;
          float vip = V[i * LDA + p], viq = V[i * LDA + q];
          V[i * LDA + p] = c * vip - s * viq;
          V[i * LDA + q] = s * vip + c * viq;
        }
      }
      __syncthreads();
      {  // row rotations: A <- J^T*A
        const int j = tid & (D - 1);
        for (int k = tid / D; k < NP; k += NG) {
          int p = pp[k], q = qq[k];
          float c = cc[k], s = ss[k];
          float apj = A[p * LDA + j], aqj = A[q * LDA + j];
          A[p * LDA + j] = c * apj - s * aqj;
          A[q * LDA + j] = s * apj + c * aqj;
        }
      }
      __syncthreads();
    }
  }

  if (tid < D) {
    float w = fmaxf(A[tid * LDA + tid], RE_EPS);
    fw[tid] = TAKELOG ? logf(w) : w;
  }
  __syncthreads();

  float* out = Sout + (size_t)b * D * D;
  for (int idx = tid; idx < D * D; idx += 256) {
    int i = idx / D, j = idx % D;
    float sum = 0.0f;
    for (int k = 0; k < D; ++k)
      sum += V[i * LDA + k] * fw[k] * V[j * LDA + k];
    out[idx] = sum;   // safe in-place: A/V fully staged in LDS
  }
}

// ---------------------------------------------------------------------------
// Kernel 4: FC head  out = feats @ fc_w^T + fc_b   (one wave per class)
// ---------------------------------------------------------------------------
__global__ __launch_bounds__(320) void fc_kernel(const float* __restrict__ feats,
                                                 const float* __restrict__ fcw,
                                                 const float* __restrict__ fcb,
                                                 float* __restrict__ out) {
  const int b    = blockIdx.x;
  const int c    = threadIdx.x >> 5;   // class 0..9
  const int lane = threadIdx.x & 31;
  const float* f = feats + (size_t)b * 1024;
  const float* w = fcw + (size_t)c * 1024;
  float sum = 0.0f;
  for (int i = lane; i < 1024; i += 32) sum += f[i] * w[i];
#pragma unroll
  for (int off = 16; off; off >>= 1) sum += __shfl_down(sum, off, 32);
  if (lane == 0) out[(size_t)b * 10 + c] = sum + fcb[c];
}

// ---------------------------------------------------------------------------
extern "C" void kernel_launch(void* const* d_in, const int* in_sizes, int n_in,
                              void* d_out, int out_size, void* d_ws, size_t ws_size,
                              hipStream_t stream) {
  (void)in_sizes; (void)n_in; (void)out_size; (void)ws_size;
  const float* x    = (const float*)d_in[0];
  const float* W1   = (const float*)d_in[1];
  const float* W2   = (const float*)d_in[2];
  const float* fc_w = (const float*)d_in[3];
  const float* fc_b = (const float*)d_in[4];
  float* out = (float*)d_out;

  float* cov = (float*)d_ws;                          // 512*128*128 f32
  float* S1  = cov + (size_t)B_DIM * C_DIM * C_DIM;   // 512*64*64  f32
  float* S2  = S1  + (size_t)B_DIM * 64 * 64;         // 512*32*32  f32

  cov_kernel<<<B_DIM, 256, 0, stream>>>(x, cov);
  bimap_kernel<128, 64><<<B_DIM, 256, 0, stream>>>(cov, W1, S1);
  jacobi_kernel<64, false><<<B_DIM, 256, 0, stream>>>(S1, S1, 10);
  bimap_kernel<64, 32><<<B_DIM, 256, 0, stream>>>(S1, W2, S2);
  jacobi_kernel<32, true><<<B_DIM, 256, 0, stream>>>(S2, S2, 12);  // ReEig+LogEig fused
  fc_kernel<<<B_DIM, 320, 0, stream>>>(S2, fc_w, fc_b, out);
}
